// MaskedSuperAttention_2757369004159
// MI455X (gfx1250) — matrix operand order repaired and checked by reading, not compiled, not run on hardware
//
#include <hip/hip_runtime.h>

// ---------------------------------------------------------------------------
// MaskedSuperAttention (agent attention) for gfx1250, bf16 WMMA pipeline with
// async global->LDS double buffering on the large GEMMs.
// Dims: B=4, N=16384, C=512, H=8, HD=64, NA=32. SCALE=1/8 folded into weights.
// Workspace ~194.4 MB.
// ---------------------------------------------------------------------------
#define NB    4
#define NTOK  16384
#define DIMC  512
#define NH    8
#define HDim  64
#define NAg   32

static const size_t OFF_XB    = 0;                      // bf16 x      67,108,864
static const size_t OFF_WCT   = 67108864;               // bf16 WcombT    524,288
static const size_t OFF_WPT   = 67633152;               // bf16 WprojT    524,288
static const size_t OFF_L     = 68157440;               // f32 logits 134,217,728
static const size_t OFF_SPART = 202375168;              // f32 partial    131,072
static const size_t OFF_S     = 202506240;              // f32 sumexp       4,096
static const size_t OFF_AXB   = 202510336;              // bf16 P1@x    1,048,576
static const size_t OFF_AV    = 203558912;              // f32 agent_v    262,144

typedef __attribute__((ext_vector_type(16))) __bf16 v16bf;
typedef __attribute__((ext_vector_type(8)))  __bf16 v8bf;
typedef __attribute__((ext_vector_type(4)))  __bf16 v4bf;
typedef __attribute__((ext_vector_type(8)))  float  v8f;
typedef __attribute__((ext_vector_type(4)))  float  v4f;

struct V16P { v8bf lo; v8bf hi; };

static __device__ __forceinline__ v16bf joinbf(v8bf lo, v8bf hi) {
  V16P t{lo, hi};
  return __builtin_bit_cast(v16bf, t);
}

// A fragment: 16x32 bf16, LDS tile row-major [M][K], stride in elements.
static __device__ __forceinline__ v16bf frag_a(const __bf16* base, int stride, int lane) {
  const int row = lane & 15, hi = lane >> 4;
  const __bf16* p = base + row * stride;
  v8bf lo = *(const v8bf*)(p + hi * 8);
  v8bf up = *(const v8bf*)(p + 16 + hi * 8);
  return joinbf(lo, up);
}

// B fragment: 32x16 bf16, staged transposed in LDS as [N][K].
static __device__ __forceinline__ v16bf frag_b(const __bf16* base, int stride, int lane) {
  const int row = lane & 15, hi = lane >> 4;
  const __bf16* p = base + row * stride + hi * 8;
  v8bf lo = *(const v8bf*)(p);
  v8bf up = *(const v8bf*)(p + 16);
  return joinbf(lo, up);
}

#define WMMA_BF16(A, B, C) \
  __builtin_amdgcn_wmma_f32_16x16x32_bf16(false, (A), false, (B), (short)0, (C), false, false)

// 16B async global->LDS copy (ASYNCcnt-tracked; CDNA5 ISA).
// Low 32 bits of a flat LDS pointer are the LDS byte offset (aperture low = 0).
static __device__ __forceinline__ void async_copy16(void* lds, const void* gptr) {
  unsigned l = (unsigned)(unsigned long long)lds;
  unsigned long long g = (unsigned long long)gptr;
  asm volatile("global_load_async_to_lds_b128 %0, %1, off" :: "v"(l), "v"(g) : "memory");
}
static __device__ __forceinline__ void wait_async_le3() {
  asm volatile("s_wait_asynccnt 0x3" ::: "memory");
}
static __device__ __forceinline__ void wait_async_le1() {
  asm volatile("s_wait_asynccnt 0x1" ::: "memory");
}
static __device__ __forceinline__ void wait_async_0() {
  asm volatile("s_wait_asynccnt 0x0" ::: "memory");
}

// ---------------------------------------------------------------------------
// Prep: WcombT[j][c] (j<256: Wk-agent logit weights, j>=256: Wq-agent), SCALE folded
// ---------------------------------------------------------------------------
__global__ __launch_bounds__(256) void k_prep_wcomb(const float* __restrict__ Wq,
                                                    const float* __restrict__ Wk,
                                                    const float* __restrict__ agent,
                                                    __bf16* __restrict__ WcT) {
  int idx = blockIdx.x * 256 + threadIdx.x;   // 262144
  int c = idx & 511, j = idx >> 9;
  const float* W = (j < 256) ? Wk : Wq;
  int jj = j & 255, h = jj >> 5, i = jj & 31;
  float acc = 0.f;
  #pragma unroll 8
  for (int d = 0; d < HDim; ++d)
    acc += W[(long)c * DIMC + h * HDim + d] * agent[i * DIMC + h * HDim + d];
  WcT[(long)j * DIMC + c] = (__bf16)(acc * 0.125f);   // SCALE = 64^-0.5
}

__global__ __launch_bounds__(256) void k_prep_wproj(const float* __restrict__ Wproj,
                                                    __bf16* __restrict__ WpT) {
  int idx = blockIdx.x * 256 + threadIdx.x;   // 262144
  int j = idx >> 9, c = idx & 511;
  WpT[(long)j * DIMC + c] = (__bf16)Wproj[(long)c * DIMC + j];
}

__global__ __launch_bounds__(256) void k_x2bf(const float* __restrict__ x,
                                              __bf16* __restrict__ xb) {
  long idx = (long)blockIdx.x * 256 + threadIdx.x;   // *4 elements
  v4f v = ((const v4f*)x)[idx];
  v4bf o;
  o[0] = (__bf16)v[0]; o[1] = (__bf16)v[1]; o[2] = (__bf16)v[2]; o[3] = (__bf16)v[3];
  ((v4bf*)xb)[idx] = o;
}

// ---------------------------------------------------------------------------
// GEMM: L[65536,512] = xb @ WcombT^T, f32 out (both stage-1 & stage-2 logits)
// Block tile 128x64; 8 waves, each a 2x2 grid of 16x16 subtiles (4 wmma/step).
// Double-buffered LDS filled by async global->LDS loads.
// ---------------------------------------------------------------------------
__global__ __launch_bounds__(256) void k_logits(const __bf16* __restrict__ xb,
                                                const __bf16* __restrict__ WcT,
                                                float* __restrict__ L) {
  __shared__ __attribute__((aligned(16))) __bf16 As[2][128 * 48];
  __shared__ __attribute__((aligned(16))) __bf16 Bs[2][64 * 48];
  const int tid = threadIdx.x, lane = tid & 31, wave = tid >> 5;
  const long mBase = (long)blockIdx.x * 128;
  const int  nBase = blockIdx.y * 64;
  const int cm = (wave & 3) * 2;        // m-subtile base (of 8)
  const int cn = (wave >> 2) * 2;       // n-subtile base (of 4)
  // staging: A row = tid>>1 (128 rows), 16 elems at col (tid&1)*16 (2x b128)
  //          B row = tid>>2 (64 rows),   8 elems at col (tid&3)*8  (1x b128)
  const int arow = tid >> 1, acol = (tid & 1) * 16;
  const int brow = tid >> 2, bcol = (tid & 3) * 8;

  auto stage = [&](int buf, int kt) {
    const int k0 = kt * 32;
    async_copy16(&As[buf][arow * 48 + acol],
                 xb + (mBase + arow) * DIMC + k0 + acol);
    async_copy16(&As[buf][arow * 48 + acol + 8],
                 xb + (mBase + arow) * DIMC + k0 + acol + 8);
    async_copy16(&Bs[buf][brow * 48 + bcol],
                 WcT + (long)(nBase + brow) * DIMC + k0 + bcol);
  };

  v8f acc00 = {}, acc01 = {}, acc10 = {}, acc11 = {};
  stage(0, 0);
  for (int kt = 0; kt < 16; ++kt) {
    const int cur = kt & 1;
    if (kt + 1 < 16) { stage(cur ^ 1, kt + 1); wait_async_le3(); }
    else             { wait_async_0(); }
    __syncthreads();
    v16bf a0 = frag_a(&As[cur][(cm * 16) * 48], 48, lane);
    v16bf a1 = frag_a(&As[cur][((cm + 1) * 16) * 48], 48, lane);
    v16bf b0 = frag_b(&Bs[cur][(cn * 16) * 48], 48, lane);
    v16bf b1 = frag_b(&Bs[cur][((cn + 1) * 16) * 48], 48, lane);
    acc00 = WMMA_BF16(a0, b0, acc00);
    acc01 = WMMA_BF16(a0, b1, acc01);
    acc10 = WMMA_BF16(a1, b0, acc10);
    acc11 = WMMA_BF16(a1, b1, acc11);
    __syncthreads();
  }
  const int col = lane & 15, hi = lane >> 4;
  #pragma unroll
  for (int r = 0; r < 8; ++r) {
    long r0 = mBase + (cm + 0) * 16 + hi * 8 + r;
    long r1 = mBase + (cm + 1) * 16 + hi * 8 + r;
    L[r0 * DIMC + nBase + (cn + 0) * 16 + col] = acc00[r];
    L[r0 * DIMC + nBase + (cn + 1) * 16 + col] = acc01[r];
    L[r1 * DIMC + nBase + (cn + 0) * 16 + col] = acc10[r];
    L[r1 * DIMC + nBase + (cn + 1) * 16 + col] = acc11[r];
  }
}

// ---------------------------------------------------------------------------
// Masked sum of exp over N for stage-1 (columns j<256), deterministic 2-stage
// ---------------------------------------------------------------------------
__global__ __launch_bounds__(256) void k_sumexp(const float* __restrict__ L,
                                                const int* __restrict__ mask,
                                                float* __restrict__ Spart) {
  const int b = blockIdx.y, chunk = blockIdx.x;   // 32 chunks of 512 rows
  const int j = threadIdx.x;                      // agent column 0..255
  float acc = 0.f;
  const long base = (long)b * NTOK + (long)chunk * 512;
  for (int r = 0; r < 512; ++r) {
    long n = base + r;
    float v = L[n * DIMC + j];
    acc += (mask[n] > 0) ? __expf(v) : 0.f;
  }
  Spart[((long)b * 32 + chunk) * 256 + j] = acc;
}

__global__ __launch_bounds__(256) void k_sreduce(const float* __restrict__ Spart,
                                                 float* __restrict__ S) {
  int idx = blockIdx.x * 256 + threadIdx.x;   // 1024
  int b = idx >> 8, j = idx & 255;
  float a = 0.f;
  #pragma unroll
  for (int ch = 0; ch < 32; ++ch) a += Spart[((long)b * 32 + ch) * 256 + j];
  S[idx] = a;
}

// ---------------------------------------------------------------------------
// Stage-1 aggregation: AXb[b, j(256), c(512)] = sum_n P1[b,j,n] * xb[b,n,c]
// A (softmax weights) generated on the fly; L/x re-reads stay L2-resident.
// ---------------------------------------------------------------------------
__global__ __launch_bounds__(256) void k_stage1(const float* __restrict__ L,
                                                const __bf16* __restrict__ xb,
                                                const int* __restrict__ mask,
                                                const float* __restrict__ S,
                                                __bf16* __restrict__ AXb) {
  __shared__ __attribute__((aligned(16))) __bf16 As[64 * 48];  // w[j][n]
  __shared__ __attribute__((aligned(16))) __bf16 Bs[64 * 48];  // x^T[c][n]
  __shared__ float ss[64];
  const int tid = threadIdx.x, lane = tid & 31, wave = tid >> 5;
  const int b = blockIdx.z;
  const int jBase = blockIdx.x * 64;
  const int cBase = blockIdx.y * 64;
  if (tid < 64) ss[tid] = 1.f / fmaxf(S[b * 256 + jBase + tid], 1e-20f);
  __syncthreads();
  const int mi = wave >> 1, ni = (wave & 1) * 2;
  const int nl = tid >> 3;         // local token 0..31
  const int js = (tid & 7) * 8;    // j / c start
  v8f acc0 = {}, acc1 = {};
  for (int kt = 0; kt < NTOK / 32; ++kt) {
    const long n = (long)b * NTOK + kt * 32 + nl;
    const float mval = (mask[n] > 0) ? 1.f : 0.f;  // masked key -> weight 0
    const float* lp = L + n * DIMC + jBase + js;
    #pragma unroll
    for (int i = 0; i < 8; ++i)
      As[(js + i) * 48 + nl] = (__bf16)(mval * __expf(lp[i]) * ss[js + i]);
    v8bf xv = *(const v8bf*)(xb + n * DIMC + cBase + js);
    #pragma unroll
    for (int i = 0; i < 8; ++i) Bs[(js + i) * 48 + nl] = xv[i];
    __syncthreads();
    v16bf af = frag_a(As + mi * 16 * 48, 48, lane);
    v16bf b0 = frag_b(Bs + ni * 16 * 48, 48, lane);
    v16bf b1 = frag_b(Bs + (ni + 1) * 16 * 48, 48, lane);
    acc0 = WMMA_BF16(af, b0, acc0);
    acc1 = WMMA_BF16(af, b1, acc1);
    __syncthreads();
  }
  const int col = lane & 15, hi = lane >> 4;
  #pragma unroll
  for (int r = 0; r < 8; ++r) {
    int j = jBase + mi * 16 + hi * 8 + r;
    AXb[((long)b * 256 + j) * DIMC + cBase + ni * 16 + col]       = (__bf16)acc0[r];
    AXb[((long)b * 256 + j) * DIMC + cBase + (ni + 1) * 16 + col] = (__bf16)acc1[r];
  }
}

// ---------------------------------------------------------------------------
// agent_v[b,h,i,d] = AXb[b, h*32+i, :] @ Wv[:, h*64+d]   (tiny: 65536 outputs)
// ---------------------------------------------------------------------------
__global__ __launch_bounds__(256) void k_agentv(const __bf16* __restrict__ AXb,
                                                const float* __restrict__ Wv,
                                                float* __restrict__ av) {
  int idx = blockIdx.x * 256 + threadIdx.x;   // 65536
  int d = idx & 63, i = (idx >> 6) & 31, h = (idx >> 11) & 7, b = idx >> 14;
  const __bf16* arow = AXb + ((long)b * 256 + h * 32 + i) * DIMC;
  float acc = 0.f;
  #pragma unroll 8
  for (int c = 0; c < DIMC; ++c)
    acc += (float)arow[c] * Wv[(long)c * DIMC + h * HDim + d];
  av[idx] = acc;   // layout [b][h][i][d]
}

// ---------------------------------------------------------------------------
// Fused stage-2 softmax + p2@agent_v -> Y (LDS, bf16), then Y @ WprojT, masked.
// Wproj tiles double-buffered via async global->LDS (prefetch starts before
// the softmax phase so DMA overlaps the VALU work).
// ---------------------------------------------------------------------------
__global__ __launch_bounds__(256) void k_out(const float* __restrict__ L,
                                             const float* __restrict__ av,
                                             const __bf16* __restrict__ WpT,
                                             const int* __restrict__ mask,
                                             float* __restrict__ out) {
  __shared__ __attribute__((aligned(16))) __bf16 Y[32 * 520];   // [tok][c], pad 8
  __shared__ __attribute__((aligned(16))) __bf16 Bs[2][64 * 48];
  __shared__ int mks[32];
  const int tid = threadIdx.x, lane = tid & 31, wave = tid >> 5;
  const long tile = blockIdx.x;                    // 2048 tiles
  const int  b = (int)(tile >> 9);                 // 512 tiles per batch
  const long nBase = (long)b * NTOK + (tile & 511) * 32;
  const int brow = tid >> 2, bcol = (tid & 3) * 8;

  auto stageB = [&](int buf, int s) {              // s = nt*16 + kt
    const int nt = s >> 4, kt = s & 15;
    async_copy16(&Bs[buf][brow * 48 + bcol],
                 WpT + (long)(nt * 64 + brow) * DIMC + kt * 32 + bcol);
  };
  stageB(0, 0);                                    // overlap with softmax phase

  if (tid < 32) mks[tid] = mask[nBase + tid];

  // ---- stage 2: per (token, head) softmax over 32 agents, y = p2 @ agent_v
  {
    const int tok = tid >> 3, h = tid & 7;         // 32 x 8 = 256 pairs
    const float* lp = L + (nBase + tok) * DIMC + 256 + h * NAg;
    float e[NAg];
    float sum = 0.f;
    #pragma unroll
    for (int i = 0; i < NAg; ++i) { e[i] = __expf(lp[i]); sum += e[i]; }
    const float inv = 1.f / sum;
    float y[HDim];
    #pragma unroll
    for (int d = 0; d < HDim; ++d) y[d] = 0.f;
    const float* ah = av + ((long)b * NH + h) * NAg * HDim;
    for (int i = 0; i < NAg; ++i) {
      const float pv = e[i] * inv;
      const float* a = ah + i * HDim;
      #pragma unroll
      for (int d = 0; d < HDim; ++d) y[d] += pv * a[d];
    }
    __bf16* yp = Y + tok * 520 + h * HDim;
    #pragma unroll
    for (int d = 0; d < HDim; ++d) yp[d] = (__bf16)y[d];
  }
  __syncthreads();

  // ---- projection: Out[32,512] = Y @ WprojT^T, masked store
  const int mi = wave >> 2, nio = wave & 3;        // 2 x 4 subtiles
  const int col = lane & 15, hi = lane >> 4;
  const v8f vzero = {};
  v8f acc = {};
  for (int s = 0; s < 128; ++s) {                  // nt = s>>4, kt = s&15
    const int nt = s >> 4, kt = s & 15;
    const int cur = s & 1;
    if (kt == 0) acc = vzero;
    if (s + 1 < 128) { stageB(cur ^ 1, s + 1); wait_async_le1(); }
    else             { wait_async_0(); }
    __syncthreads();
    v16bf af = frag_a(Y + (mi * 16) * 520 + kt * 32, 520, lane);
    v16bf bf = frag_b(&Bs[cur][nio * 16 * 48], 48, lane);
    acc = WMMA_BF16(af, bf, acc);
    __syncthreads();
    if (kt == 15) {
      #pragma unroll
      for (int r = 0; r < 8; ++r) {
        const int tok = mi * 16 + hi * 8 + r;
        const float sm = (mks[tok] > 0) ? 1.f : 0.f;
        out[(nBase + tok) * DIMC + nt * 64 + nio * 16 + col] = sm * acc[r];
      }
    }
  }
}

// ---------------------------------------------------------------------------
extern "C" void kernel_launch(void* const* d_in, const int* in_sizes, int n_in,
                              void* d_out, int out_size, void* d_ws, size_t ws_size,
                              hipStream_t stream) {
  (void)in_sizes; (void)n_in; (void)out_size; (void)ws_size;
  const float* x     = (const float*)d_in[0];
  const int*   mask  = (const int*)d_in[1];
  const float* agent = (const float*)d_in[2];
  const float* Wq    = (const float*)d_in[3];
  const float* Wk    = (const float*)d_in[4];
  const float* Wv    = (const float*)d_in[5];
  const float* Wproj = (const float*)d_in[6];
  float* out = (float*)d_out;

  char* ws = (char*)d_ws;
  __bf16* xb    = (__bf16*)(ws + OFF_XB);
  __bf16* WcT   = (__bf16*)(ws + OFF_WCT);
  __bf16* WpT   = (__bf16*)(ws + OFF_WPT);
  float*  L     = (float*)(ws + OFF_L);
  float*  Spart = (float*)(ws + OFF_SPART);
  float*  S     = (float*)(ws + OFF_S);
  __bf16* AXb   = (__bf16*)(ws + OFF_AXB);
  float*  av    = (float*)(ws + OFF_AV);

  k_prep_wcomb<<<1024, 256, 0, stream>>>(Wq, Wk, agent, WcT);
  k_prep_wproj<<<1024, 256, 0, stream>>>(Wproj, WpT);
  k_x2bf<<<(NB * NTOK * DIMC) / (256 * 4), 256, 0, stream>>>(x, xb);

  k_logits<<<dim3((NB * NTOK) / 128, DIMC / 64), 256, 0, stream>>>(xb, WcT, L);

  k_sumexp<<<dim3(32, NB), 256, 0, stream>>>(L, mask, Spart);
  k_sreduce<<<NB, 256, 0, stream>>>(Spart, S);

  k_stage1<<<dim3(4, 8, NB), 256, 0, stream>>>(L, xb, mask, S, AXb);
  k_agentv<<<(NB * NH * NAg * HDim) / 256, 256, 0, stream>>>(AXb, Wv, av);

  k_out<<<(NB * NTOK) / 32, 256, 0, stream>>>(L, av, WpT, mask, out);
}
